// ProtoNet_67310727463516
// MI455X (gfx1250) — matrix operand order repaired and checked by reading, not compiled, not run on hardware
//
#include <hip/hip_runtime.h>

#define C_CLS 256
#define S_SUP 256
#define D_IN  2048
#define H_DIM 2048
#define O_DIM 2048

typedef __attribute__((ext_vector_type(16))) __bf16 v16bf;
typedef __attribute__((ext_vector_type(8)))  __bf16 v8bf;
typedef __attribute__((ext_vector_type(4)))  __bf16 v4bf;
typedef __attribute__((ext_vector_type(8)))  float  v8f;

// f32 -> bf16, round-to-nearest-even
__device__ __forceinline__ __bf16 f2bf(float f) {
  union { float f; unsigned u; } x; x.f = f;
  unsigned r = x.u + 0x7FFFu + ((x.u >> 16) & 1u);
  unsigned short h = (unsigned short)(r >> 16);
  return __builtin_bit_cast(__bf16, h);
}

// A fragment (16x32 bf16): lane holds K = kg*8..kg*8+7 and 16+kg*8..+7
// caller passes p = &row_base[kg*8]
__device__ __forceinline__ v16bf load_frag_a(const __bf16* p) {
  v8bf lo = *(const v8bf*)(p);
  v8bf hi = *(const v8bf*)(p + 16);
  v16bf r;
#pragma unroll
  for (int i = 0; i < 8; ++i) { r[i] = lo[i]; r[8 + i] = hi[i]; }
  return r;
}

// B fragment (32x16 bf16): lane holds K = kg*8..kg*8+7 and 16+kg*8..+7 of column n
// caller passes p = &row_base[kg*8]
__device__ __forceinline__ v16bf load_frag_b(const __bf16* p) {
  v8bf lo = *(const v8bf*)(p);
  v8bf hi = *(const v8bf*)(p + 16);
  v16bf r;
#pragma unroll
  for (int i = 0; i < 8; ++i) { r[i] = lo[i]; r[8 + i] = hi[i]; }
  return r;
}

// ---------------- Phase 1: prototypes = mean over s ----------------
__global__ __launch_bounds__(256)
void proto_kernel(const float* __restrict__ support, float* __restrict__ proto) {
  const int c = blockIdx.x;
  const int t = threadIdx.x;
  const float* base = support + (size_t)c * S_SUP * D_IN;
#pragma unroll
  for (int i = 0; i < D_IN / 256; ++i) {
    const int d = t + i * 256;
    float acc = 0.f;
    for (int s = 0; s < S_SUP; ++s) acc += base[(size_t)s * D_IN + d];
    proto[(size_t)c * D_IN + d] = acc * (1.0f / S_SUP);
  }
}

// ------- Phases 2 & 5: Out(MxN) = [relu](A(MxK) * W(NxK)^T + bias) ----------
template <bool RELU>
__global__ __launch_bounds__(256)
void gemm_nt_bias(const float* __restrict__ A, const float* __restrict__ W,
                  const float* __restrict__ bias, float* __restrict__ Out,
                  int N, int K) {
  constexpr int LDSK = 40;                      // 32 + 8 pad (80B rows, 16B aligned)
  __shared__ __bf16 As[128 * LDSK];
  __shared__ __bf16 Bs[128 * LDSK];
  const int t = threadIdx.x;
  const int lane = t & 31;
  const int w = t >> 5;
  const int wrow = w & 1;                       // 2 strips of 64 rows
  const int wcol = w >> 1;                      // 4 strips of 32 cols
  const int mbase = blockIdx.y * 128;
  const int nbase = blockIdx.x * 128;
  const int lrow = lane & 15;
  const int lkg  = lane >> 4;

  const v8f zero = {0.f,0.f,0.f,0.f,0.f,0.f,0.f,0.f};
  v8f acc[4][2];
#pragma unroll
  for (int i = 0; i < 4; ++i)
#pragma unroll
    for (int j = 0; j < 2; ++j) acc[i][j] = zero;

  for (int k0 = 0; k0 < K; k0 += 32) {
#pragma unroll
    for (int p = 0; p < 4; ++p) {               // 128x32 f32 -> bf16 tiles
      const int e = p * 1024 + t * 4;
      const int r = e >> 5, col = e & 31;
      const float4 va = *(const float4*)(A + (size_t)(mbase + r) * K + k0 + col);
      v4bf pa = { f2bf(va.x), f2bf(va.y), f2bf(va.z), f2bf(va.w) };
      *(v4bf*)(&As[r * LDSK + col]) = pa;
      const float4 vb = *(const float4*)(W + (size_t)(nbase + r) * K + k0 + col);
      v4bf pb = { f2bf(vb.x), f2bf(vb.y), f2bf(vb.z), f2bf(vb.w) };
      *(v4bf*)(&Bs[r * LDSK + col]) = pb;
    }
    __syncthreads();
    v16bf af[4], bfr[2];
#pragma unroll
    for (int mt = 0; mt < 4; ++mt)
      af[mt] = load_frag_a(&As[(wrow * 64 + mt * 16 + lrow) * LDSK + lkg * 8]);
#pragma unroll
    for (int nt = 0; nt < 2; ++nt)
      bfr[nt] = load_frag_b(&Bs[(wcol * 32 + nt * 16 + lrow) * LDSK + lkg * 8]);
#pragma unroll
    for (int mt = 0; mt < 4; ++mt)
#pragma unroll
      for (int nt = 0; nt < 2; ++nt)
        acc[mt][nt] = __builtin_amdgcn_wmma_f32_16x16x32_bf16(
            false, af[mt], false, bfr[nt], (short)0, acc[mt][nt], false, false);
    __syncthreads();
  }
#pragma unroll
  for (int nt = 0; nt < 2; ++nt) {
    const int n = nbase + wcol * 32 + nt * 16 + lrow;
    const float bn = bias[n];
#pragma unroll
    for (int mt = 0; mt < 4; ++mt)
#pragma unroll
      for (int j = 0; j < 8; ++j) {
        const int m = mbase + wrow * 64 + mt * 16 + (lkg << 3) + j;
        float v = acc[mt][nt][j] + bn;
        if (RELU) v = fmaxf(v, 0.0f);
        Out[(size_t)m * N + n] = v;
      }
  }
}

// -------- Phase 3: -||proto - q_h||^2 then row softmax -> probs ---------
__global__ __launch_bounds__(256)
void dist_softmax_kernel(const float* __restrict__ proto, const float* __restrict__ qh,
                         float* __restrict__ probs) {
  __shared__ float qrow[D_IN];
  __shared__ float dist[C_CLS];
  __shared__ float redm[8];
  __shared__ float reds[8];
  const int q = blockIdx.x;
  const int t = threadIdx.x;
  const int lane = t & 31;
  const int w = t >> 5;
#pragma unroll
  for (int i = 0; i < D_IN / 256; ++i)
    qrow[t + i * 256] = qh[(size_t)q * D_IN + t + i * 256];
  __syncthreads();
  for (int i = 0; i < 32; ++i) {                 // each wave handles 32 classes
    const int cc = w * 32 + i;
    const float* prow = proto + (size_t)cc * D_IN;
    float s = 0.f;
    for (int d = lane; d < D_IN; d += 32) {
      const float df = prow[d] - qrow[d];
      s += df * df;
    }
#pragma unroll
    for (int off = 16; off > 0; off >>= 1) s += __shfl_xor(s, off, 32);
    if (lane == 0) dist[cc] = -s;
  }
  __syncthreads();
  const float v = dist[t];
  float m = v;
#pragma unroll
  for (int off = 16; off > 0; off >>= 1) m = fmaxf(m, __shfl_xor(m, off, 32));
  if (lane == 0) redm[w] = m;
  __syncthreads();
  float bm = redm[0];
#pragma unroll
  for (int i = 1; i < 8; ++i) bm = fmaxf(bm, redm[i]);
  const float e = __expf(v - bm);
  float s = e;
#pragma unroll
  for (int off = 16; off > 0; off >>= 1) s += __shfl_xor(s, off, 32);
  if (lane == 0) reds[w] = s;
  __syncthreads();
  float tot = 0.f;
#pragma unroll
  for (int i = 0; i < 8; ++i) tot += reds[i];
  probs[(size_t)q * C_CLS + t] = e / tot;
}

// -------- Phase 4 (dominant): wh[c,h] = sum_s probs[c,s]*relu(support[c,s,:].W_h[h,:]+b_h[h])
// One WG per (h-tile of 128, class). Tile: 256(s) x 128(h), K-step 32, bf16 WMMA.
__global__ __launch_bounds__(256)
void fused_support_hidden(const float* __restrict__ support,
                          const float* __restrict__ W_h,
                          const float* __restrict__ b_h,
                          const float* __restrict__ probs,
                          float* __restrict__ wh) {
  constexpr int LDSK = 40;
  __shared__ __bf16 As[256 * LDSK];              // support tile (s x k)
  __shared__ __bf16 Bs[128 * LDSK];              // W_h tile (h x k)
  __shared__ float pr[S_SUP];
  __shared__ float red[2][4][64];
  const int ht = blockIdx.x;                     // 16 h-tiles
  const int c  = blockIdx.y;                     // 256 classes
  const int t = threadIdx.x;
  const int lane = t & 31;
  const int w = t >> 5;
  const int wrow = w & 3;                        // 4 strips of 64 s-rows
  const int wcol = w >> 2;                       // 2 strips of 64 h-cols
  const int lrow = lane & 15;
  const int lkg  = lane >> 4;

  pr[t] = probs[(size_t)c * S_SUP + t];

  const float* Abase = support + (size_t)c * S_SUP * D_IN;
  const float* Bbase = W_h + (size_t)ht * 128 * D_IN;

  const v8f zero = {0.f,0.f,0.f,0.f,0.f,0.f,0.f,0.f};
  v8f acc[4][4];
#pragma unroll
  for (int i = 0; i < 4; ++i)
#pragma unroll
    for (int j = 0; j < 4; ++j) acc[i][j] = zero;

  for (int k0 = 0; k0 < D_IN; k0 += 32) {
#pragma unroll
    for (int p = 0; p < 8; ++p) {                // A: 256x32 (8 passes), B: 128x32 (4)
      const int e = p * 1024 + t * 4;
      const int r = e >> 5, col = e & 31;
      const float4 va = *(const float4*)(Abase + (size_t)r * D_IN + k0 + col);
      v4bf pa = { f2bf(va.x), f2bf(va.y), f2bf(va.z), f2bf(va.w) };
      *(v4bf*)(&As[r * LDSK + col]) = pa;
      if (p < 4) {
        const float4 vb = *(const float4*)(Bbase + (size_t)r * D_IN + k0 + col);
        v4bf pb = { f2bf(vb.x), f2bf(vb.y), f2bf(vb.z), f2bf(vb.w) };
        *(v4bf*)(&Bs[r * LDSK + col]) = pb;
      }
    }
    __syncthreads();
    v16bf af[4], bfr[4];
#pragma unroll
    for (int mt = 0; mt < 4; ++mt)
      af[mt] = load_frag_a(&As[(wrow * 64 + mt * 16 + lrow) * LDSK + lkg * 8]);
#pragma unroll
    for (int nt = 0; nt < 4; ++nt)
      bfr[nt] = load_frag_b(&Bs[(wcol * 64 + nt * 16 + lrow) * LDSK + lkg * 8]);
#pragma unroll
    for (int mt = 0; mt < 4; ++mt)
#pragma unroll
      for (int nt = 0; nt < 4; ++nt)
        acc[mt][nt] = __builtin_amdgcn_wmma_f32_16x16x32_bf16(
            false, af[mt], false, bfr[nt], (short)0, acc[mt][nt], false, false);
    __syncthreads();
  }

  // Epilogue: +b_h, relu, weight by probs[c,s], reduce over s (no atomics).
  float part[4] = {0.f, 0.f, 0.f, 0.f};
#pragma unroll
  for (int nt = 0; nt < 4; ++nt) {
    const int hloc = wcol * 64 + nt * 16 + lrow;
    const float bh = b_h[ht * 128 + hloc];
#pragma unroll
    for (int mt = 0; mt < 4; ++mt)
#pragma unroll
      for (int j = 0; j < 8; ++j) {
        const int s = wrow * 64 + mt * 16 + (lkg << 3) + j;   // C-layout: VGPR j -> M=8*(lane/16)+j
        const float v = fmaxf(acc[mt][nt][j] + bh, 0.0f);
        part[nt] += v * pr[s];
      }
  }
#pragma unroll
  for (int nt = 0; nt < 4; ++nt)                 // lanes L and L+16 share h
    part[nt] += __shfl_xor(part[nt], 16, 32);
  if (lane < 16) {
#pragma unroll
    for (int nt = 0; nt < 4; ++nt)
      red[wcol][wrow][nt * 16 + lane] = part[nt];
  }
  __syncthreads();
  if (t < 128) {
    const int wc = t >> 6, hx = t & 63;
    const float ssum = red[wc][0][hx] + red[wc][1][hx] + red[wc][2][hx] + red[wc][3][hx];
    wh[(size_t)c * H_DIM + ht * 128 + t] = ssum;
  }
}

extern "C" void kernel_launch(void* const* d_in, const int* in_sizes, int n_in,
                              void* d_out, int out_size, void* d_ws, size_t ws_size,
                              hipStream_t stream) {
  (void)in_sizes; (void)n_in; (void)out_size; (void)ws_size;
  const float* support = (const float*)d_in[0];   // (256,256,2048)
  const float* query   = (const float*)d_in[1];   // (256,2048)
  const float* W_h     = (const float*)d_in[2];   // (2048,2048)
  const float* b_h     = (const float*)d_in[3];   // (2048)
  const float* W_o     = (const float*)d_in[4];   // (2048,2048)
  const float* b_o     = (const float*)d_in[5];   // (2048)
  float* out = (float*)d_out;                     // (256,2048)

  float* ws    = (float*)d_ws;
  float* proto = ws;                              // 256*2048
  float* qh    = proto + (size_t)C_CLS * D_IN;    // 256*2048
  float* probs = qh + (size_t)C_CLS * H_DIM;      // 256*256
  float* wh    = probs + (size_t)C_CLS * C_CLS;   // 256*2048

  proto_kernel<<<C_CLS, 256, 0, stream>>>(support, proto);
  gemm_nt_bias<true><<<dim3(H_DIM / 128, C_CLS / 128), 256, 0, stream>>>(
      query, W_h, b_h, qh, H_DIM, D_IN);
  dist_softmax_kernel<<<C_CLS, 256, 0, stream>>>(proto, qh, probs);
  fused_support_hidden<<<dim3(H_DIM / 128, C_CLS), 256, 0, stream>>>(
      support, W_h, b_h, probs, wh);
  gemm_nt_bias<false><<<dim3(O_DIM / 128, C_CLS / 128), 256, 0, stream>>>(
      wh, W_o, b_o, out, O_DIM, H_DIM);
}
